// WindowAttention_26379689132414
// MI455X (gfx1250) — matrix operand order repaired and checked by reading, not compiled, not run on hardware
//
#include <hip/hip_runtime.h>
#include <hip/hip_bf16.h>
#include <math.h>

// ---------------------------------------------------------------------------
// WindowAttention for MI455X (gfx1250, wave32, WMMA).
// Shapes: B=4, N=1024, C=256, heads=8, d=32, N8=128.
// Big GEMMs (y/Q/K projections, Q*K^T) use v_wmma_f32_16x16x32_f16 (fp32 acc).
// dots + pair-bias + double-softmax fused: 16x1024 score tile lives in LDS.
// Pair bias precomputed once (p_r read once, non-temporal: 50MB) into 16MB ws
// that stays L2-resident across the 8 head passes; the 134MB attention output
// is streamed with non-temporal stores so it does not evict the bias from L2.
// xv/Wv path is dead code in the reference: skipped.
// ---------------------------------------------------------------------------

typedef __attribute__((ext_vector_type(16))) _Float16 v16h;
typedef __attribute__((ext_vector_type(8)))  float    v8f;

#define EPS_BN 1e-5f
#define NN 1024
#define CC 256
#define HH 8
#define DD 32
#define NQ8 128

// A fragment: 16x32 f16 (rows m0..m0+15, cols kb..kb+31) from row-major f32.
// ISA layout: lanes 0-15 row M=lane hold K={0..7,16..23}; lanes 16-31 hold
// K={8..15,24..31}. Converted f32->f16 on the fly.
__device__ __forceinline__ v16h load_fragA_f32(const float* __restrict__ A,
                                               int lda, int m0, int kb, int lane) {
  const int m = m0 + (lane & 15);
  const float* p = A + (size_t)m * lda + kb + ((lane & 16) ? 8 : 0);
  v16h f;
#pragma unroll
  for (int i = 0; i < 8; ++i) f[i] = (_Float16)p[i];
#pragma unroll
  for (int i = 0; i < 8; ++i) f[i + 8] = (_Float16)p[16 + i];
  return f;
}

// B fragment: 32x16 f16 with B[k][n] = W[n0+n][kb+k] (W row-major N x K, i.e.
// computing X @ W^T). ISA layout: lanes 0-15 col N=lane hold K={0..7,16..23},
// lanes 16-31 hold K={8..15,24..31} (same order as the A fragment).
__device__ __forceinline__ v16h load_fragB_rowT(const float* __restrict__ W,
                                                int ldw, int n0, int kb, int lane) {
  const int n = n0 + (lane & 15);
  const float* p = W + (size_t)n * ldw + kb + ((lane & 16) ? 8 : 0);
  v16h f;
#pragma unroll
  for (int i = 0; i < 8; ++i) f[i] = (_Float16)p[i];
#pragma unroll
  for (int i = 0; i < 8; ++i) f[i + 8] = (_Float16)p[16 + i];
  return f;
}

// ---------------------------------------------------------------------------
// C[m,n] = act(bn(sum_k A[m,k]*W[n,k] + bias[n]))   (fp32 in, f16 WMMA, fp32 out)
// Block: 128 threads = 4 waves; block tile 64x64; wave tile 16x64 (4 WMMA accs).
// ---------------------------------------------------------------------------
__global__ __launch_bounds__(128)
void wmma_gemm_xwT(const float* __restrict__ A, const float* __restrict__ W,
                   const float* __restrict__ bias,
                   const float* __restrict__ bn_g, const float* __restrict__ bn_b,
                   const float* __restrict__ bn_m, const float* __restrict__ bn_v,
                   int fuse_bn_relu, float* __restrict__ C, int M, int N, int K) {
  const int lane = threadIdx.x & 31;
  const int wv   = threadIdx.x >> 5;
  const int m0 = blockIdx.x * 64 + wv * 16;
  const int n0 = blockIdx.y * 64;

  v8f acc[4] = {};
  for (int kb = 0; kb < K; kb += 32) {
    v16h a = load_fragA_f32(A, K, m0, kb, lane);
#pragma unroll
    for (int t = 0; t < 4; ++t) {
      v16h b = load_fragB_rowT(W, K, n0 + t * 16, kb, lane);
      acc[t] = __builtin_amdgcn_wmma_f32_16x16x32_f16(
          false, a, false, b, (short)0, acc[t], false, false);
    }
  }

  // C/D layout: element r -> row M = r (lanes 0-15) or r+8 (lanes 16-31),
  // col N = lane & 15.
  const int mro  = (lane >> 4) << 3;
  const int ncol = lane & 15;
#pragma unroll
  for (int t = 0; t < 4; ++t) {
    const int n = n0 + t * 16 + ncol;
    float sc = 1.f, sh = 0.f;
    if (fuse_bn_relu) {
      sc = bn_g[n] * rsqrtf(bn_v[n] + EPS_BN);
      sh = bn_b[n] - bn_m[n] * sc;
    }
    const float bs = bias[n];
#pragma unroll
    for (int r = 0; r < 8; ++r) {
      const int m = m0 + mro + r;
      float v = acc[t][r] + bs;
      if (fuse_bn_relu) v = fmaxf(v * sc + sh, 0.f);
      C[(size_t)m * N + n] = v;
    }
  }
}

// ---------------------------------------------------------------------------
// Pair bias: bias[b,i,j] = Wp2 . relu(bn_p(p_r[b,0,i,j,:] @ Wp1^T + bp1)) + bp2
// p_r is read exactly once -> non-temporal loads (don't pollute L2).
// The 16MB bias result is re-read 8x (per head) -> regular (RT) stores.
// ---------------------------------------------------------------------------
__global__ __launch_bounds__(256)
void bias_kernel(const float* __restrict__ p_r,
                 const float* __restrict__ Wp1, const float* __restrict__ bp1,
                 const float* __restrict__ g,  const float* __restrict__ bb,
                 const float* __restrict__ mm, const float* __restrict__ vv,
                 const float* __restrict__ Wp2, const float* __restrict__ bp2,
                 float* __restrict__ biasw, int total) {
  const int idx = blockIdx.x * blockDim.x + threadIdx.x;
  if (idx >= total) return;
  const float p0 = __builtin_nontemporal_load(p_r + (size_t)idx * 3 + 0);
  const float p1 = __builtin_nontemporal_load(p_r + (size_t)idx * 3 + 1);
  const float p2 = __builtin_nontemporal_load(p_r + (size_t)idx * 3 + 2);
  float acc = bp2[0];
#pragma unroll
  for (int c = 0; c < 3; ++c) {
    float t = Wp1[c * 3 + 0] * p0 + Wp1[c * 3 + 1] * p1 + Wp1[c * 3 + 2] * p2 + bp1[c];
    t = (t - mm[c]) * (g[c] * rsqrtf(vv[c] + EPS_BN)) + bb[c];
    acc += Wp2[c] * fmaxf(t, 0.f);
  }
  biasw[idx] = acc;
}

// ---------------------------------------------------------------------------
// Fused attention: per block -> one (b, h, 16-row query tile).
// 8 waves * 8 WMMA(16x16x32, K=d=32) cover the full 16x1024 score tile in LDS.
// Bias add folded into the softmax max-pass (coalesced b32 reads of the
// L2-resident bias). Attention output streamed with non-temporal stores.
// ---------------------------------------------------------------------------
__global__ __launch_bounds__(256)
void attn_softmax_kernel(const float* __restrict__ Q, const float* __restrict__ Km,
                         const float* __restrict__ biasw, float* __restrict__ attn) {
  __shared__ float tile[16][NN];  // 64KB, well under 320KB/WGP
  const int lane = threadIdx.x & 31;
  const int wv   = threadIdx.x >> 5;   // 0..7
  const int i0 = blockIdx.x * 16;      // query row tile
  const int h  = blockIdx.y;
  const int b  = blockIdx.z;
  const float scale = 0.17677669529663687f; // 32^-0.5
  const float* Qb = Q  + (size_t)b * NN * CC;
  const float* Kb = Km + (size_t)b * NN * CC;

  const v16h a = load_fragA_f32(Qb, CC, i0, h * DD, lane); // 16 rows x d=32
  const v8f  zero = {};
  const int mro  = (lane >> 4) << 3;
  const int ncol = lane & 15;
#pragma unroll
  for (int jt = 0; jt < 8; ++jt) {
    const int j0 = (wv * 8 + jt) * 16;
    v16h bf = load_fragB_rowT(Kb, CC, j0, h * DD, lane); // B[k][n] = K[j0+n][hd+k]
    v8f acc = __builtin_amdgcn_wmma_f32_16x16x32_f16(
        false, a, false, bf, (short)0, zero, false, false);
#pragma unroll
    for (int r = 0; r < 8; ++r)
      tile[r + mro][j0 + ncol] = acc[r] * scale;
  }
  __syncthreads();

  // Row softmax: 16 threads per row, shfl-xor reductions at width 16 (wave32).
  const int row = threadIdx.x >> 4;
  const int sub = threadIdx.x & 15;
  const float* bp = biasw + ((size_t)b * NN + (i0 + row)) * NN;  // coalesced
  float mx = -1e30f;
  for (int c = sub; c < NN; c += 16) {
    const float v = tile[row][c] + bp[c];
    tile[row][c] = v;
    mx = fmaxf(mx, v);
  }
#pragma unroll
  for (int d = 8; d >= 1; d >>= 1) mx = fmaxf(mx, __shfl_xor(mx, d, 16));
  float s = 0.f;
  for (int c = sub; c < NN; c += 16) {
    float e = __expf(tile[row][c] - mx);
    tile[row][c] = e;
    s += e;
  }
#pragma unroll
  for (int d = 8; d >= 1; d >>= 1) s += __shfl_xor(s, d, 16);
  const float inv = 1.f / s;
  const float sc  = inv / (1e-9f + s * inv);  // attn / (1e-9 + attn.sum())
  float* op = attn + (((size_t)(b * HH + h)) * NN + (i0 + row)) * NN;
  for (int c = sub; c < NN; c += 16)
    __builtin_nontemporal_store(tile[row][c] * sc, op + c);  // 134MB stream, TH=NT
}

// ---------------------------------------------------------------------------
// Context branch (tiny FLOPs): pq/pk, sim->aff softmax, pv, final out.
// ---------------------------------------------------------------------------
__global__ __launch_bounds__(256)
void pqk_kernel(const float* __restrict__ y,
                const float* __restrict__ Wqc, const float* __restrict__ Wkc,
                const float* __restrict__ g1, const float* __restrict__ b1,
                const float* __restrict__ m1, const float* __restrict__ v1,
                const float* __restrict__ g2, const float* __restrict__ b2,
                const float* __restrict__ m2, const float* __restrict__ v2,
                float* __restrict__ pq, float* __restrict__ pk) {
  const int idx = blockIdx.x * blockDim.x + threadIdx.x;  // 4*128*256
  const int c = idx & 255;
  const int q = (idx >> 8) & 127;
  const int b = idx >> 15;
  const float* yb = y + (size_t)b * NN * CC + c;
  const float* wq = Wqc + (size_t)q * NN;
  const float* wk = Wkc + (size_t)q * NN;
  float sq = 0.f, sk = 0.f;
  for (int n = 0; n < NN; ++n) {
    const float xv = yb[(size_t)n * CC];
    sq += wq[n] * xv;
    sk += wk[n] * xv;
  }
  pq[idx] = fmaxf((sq - m1[q]) * (g1[q] * rsqrtf(v1[q] + EPS_BN)) + b1[q], 0.f);
  pk[idx] = fmaxf((sk - m2[q]) * (g2[q] * rsqrtf(v2[q] + EPS_BN)) + b2[q], 0.f);
}

__global__ __launch_bounds__(256)
void simaff_kernel(const float* __restrict__ pq, const float* __restrict__ pk,
                   float* __restrict__ aff) {
  __shared__ float red[256];
  const int e = threadIdx.x;
  const int c = blockIdx.x & 255;
  const int b = blockIdx.x >> 8;
  const float* pkb = pk + (size_t)b * NQ8 * CC;
  const float* pqb = pq + (size_t)b * NQ8 * CC;
  float s = 0.f;
  for (int q = 0; q < NQ8; ++q) s += pkb[q * CC + c] * pqb[q * CC + e];

  red[e] = s; __syncthreads();
  for (int off = 128; off > 0; off >>= 1) {
    if (e < off) red[e] = fmaxf(red[e], red[e + off]);
    __syncthreads();
  }
  const float M0 = red[0]; __syncthreads();
  const float z = M0 - s;                         // softmax(max - sim)
  red[e] = z; __syncthreads();
  for (int off = 128; off > 0; off >>= 1) {
    if (e < off) red[e] = fmaxf(red[e], red[e + off]);
    __syncthreads();
  }
  const float zm = red[0]; __syncthreads();
  const float ez = __expf(z - zm);
  red[e] = ez; __syncthreads();
  for (int off = 128; off > 0; off >>= 1) {
    if (e < off) red[e] += red[e + off];
    __syncthreads();
  }
  aff[(size_t)blockIdx.x * CC + e] = ez / red[0];
}

__global__ __launch_bounds__(256)
void pv_kernel(const float* __restrict__ y, const float* __restrict__ Wvc,
               const float* __restrict__ g,  const float* __restrict__ bb,
               const float* __restrict__ mm, const float* __restrict__ vv,
               float* __restrict__ pv) {
  const int idx = blockIdx.x * blockDim.x + threadIdx.x;  // 4*256*1024
  const int n = idx & 1023;
  const int e = (idx >> 10) & 255;
  const int b = idx >> 18;
  const float* yb = y + (size_t)b * NN * CC + (size_t)n * CC;
  const float* w  = Wvc + (size_t)e * CC;
  float s = 0.f;
  for (int c = 0; c < CC; ++c) s += w[c] * yb[c];
  pv[idx] = fmaxf((s - mm[e]) * (g[e] * rsqrtf(vv[e] + EPS_BN)) + bb[e], 0.f);
}

__global__ __launch_bounds__(256)
void out_kernel(const float* __restrict__ aff, const float* __restrict__ pv,
                const float* __restrict__ y, const float* __restrict__ alpha,
                float* __restrict__ out) {
  const int idx = blockIdx.x * blockDim.x + threadIdx.x;  // [b][n][c], 4*1024*256
  const int c = idx & 255;
  const int n = (idx >> 8) & 1023;
  const int b = idx >> 18;
  const float* ab = aff + ((size_t)b * CC + c) * CC;
  const float* pb = pv + (size_t)b * CC * NN + n;
  float s = 0.f;
  for (int e = 0; e < CC; ++e) s += ab[e] * pb[(size_t)e * NN];
  out[idx] = alpha[0] * s + y[idx];  // shortcut is y (swapped twice)
}

// ---------------------------------------------------------------------------
// Input flattening (setup_inputs dict order, params/bn dicts in insertion order):
//  0:x 1:p_r 2:W_li 3:b_li 4-7:bn_li{g,b,m,v} 8:Wq 9:bq 10:Wk 11:bk 12:Wv 13:bv
//  14:Wp1 15:bp1 16-19:bn_p{g,b,m,v} 20:Wp2 21:bp2 22:Wqc 23-26:bn1 27:Wkc
//  28-31:bn2 32:Wvc 33-36:bn3 37:alpha
// ---------------------------------------------------------------------------
extern "C" void kernel_launch(void* const* d_in, const int* in_sizes, int n_in,
                              void* d_out, int out_size, void* d_ws, size_t ws_size,
                              hipStream_t stream) {
  const float* x     = (const float*)d_in[0];
  const float* p_r   = (const float*)d_in[1];
  const float* W_li  = (const float*)d_in[2];
  const float* b_li  = (const float*)d_in[3];
  const float* bli_g = (const float*)d_in[4];
  const float* bli_b = (const float*)d_in[5];
  const float* bli_m = (const float*)d_in[6];
  const float* bli_v = (const float*)d_in[7];
  const float* Wq    = (const float*)d_in[8];
  const float* bq    = (const float*)d_in[9];
  const float* Wk    = (const float*)d_in[10];
  const float* bk    = (const float*)d_in[11];
  // d_in[12]=Wv, d_in[13]=bv: dead in the reference (xv unused)
  const float* Wp1   = (const float*)d_in[14];
  const float* bp1   = (const float*)d_in[15];
  const float* bnp_g = (const float*)d_in[16];
  const float* bnp_b = (const float*)d_in[17];
  const float* bnp_m = (const float*)d_in[18];
  const float* bnp_v = (const float*)d_in[19];
  const float* Wp2   = (const float*)d_in[20];
  const float* bp2   = (const float*)d_in[21];
  const float* Wqc   = (const float*)d_in[22];
  const float* b1_g  = (const float*)d_in[23];
  const float* b1_b  = (const float*)d_in[24];
  const float* b1_m  = (const float*)d_in[25];
  const float* b1_v  = (const float*)d_in[26];
  const float* Wkc   = (const float*)d_in[27];
  const float* b2_g  = (const float*)d_in[28];
  const float* b2_b  = (const float*)d_in[29];
  const float* b2_m  = (const float*)d_in[30];
  const float* b2_v  = (const float*)d_in[31];
  const float* Wvc   = (const float*)d_in[32];
  const float* b3_g  = (const float*)d_in[33];
  const float* b3_b  = (const float*)d_in[34];
  const float* b3_m  = (const float*)d_in[35];
  const float* b3_v  = (const float*)d_in[36];
  const float* alpha = (const float*)d_in[37];

  // Workspace layout (floats), ~34MB total
  float* ws   = (float*)d_ws;
  float* y    = ws;                 // 4*1024*256      = 1048576
  float* q    = y   + 1048576;      // 1048576
  float* k    = q   + 1048576;      // 1048576
  float* bw   = k   + 1048576;      // 4*1024*1024     = 4194304
  float* pq   = bw  + 4194304;      // 4*128*256       = 131072
  float* pk   = pq  + 131072;       // 131072
  float* aff  = pk  + 131072;       // 4*256*256       = 262144
  float* pv   = aff + 262144;       // 4*256*1024      = 1048576

  float* out  = (float*)d_out;           // (4,1024,256)
  float* attn = out + 4 * 1024 * 256;    // (4,8,1,1024,1024)

  const dim3 gg(64, 4, 1);  // M=4096/64, N=256/64
  // y = relu(bn(x @ W_li^T + b_li))
  wmma_gemm_xwT<<<gg, 128, 0, stream>>>(x, W_li, b_li, bli_g, bli_b, bli_m, bli_v,
                                        1, y, 4096, 256, 256);
  // Q = y @ Wq^T + bq ; K = y @ Wk^T + bk
  wmma_gemm_xwT<<<gg, 128, 0, stream>>>(y, Wq, bq, nullptr, nullptr, nullptr, nullptr,
                                        0, q, 4096, 256, 256);
  wmma_gemm_xwT<<<gg, 128, 0, stream>>>(y, Wk, bk, nullptr, nullptr, nullptr, nullptr,
                                        0, k, 4096, 256, 256);
  // pair bias (p_r read once, NT)
  bias_kernel<<<16384, 256, 0, stream>>>(p_r, Wp1, bp1, bnp_g, bnp_b, bnp_m, bnp_v,
                                         Wp2, bp2, bw, 4 * 1024 * 1024);
  // fused dots + bias + softmax + renorm -> attention output (NT stores)
  attn_softmax_kernel<<<dim3(64, HH, 4), 256, 0, stream>>>(q, k, bw, attn);
  // context branch
  pqk_kernel<<<512, 256, 0, stream>>>(y, Wqc, Wkc, b1_g, b1_b, b1_m, b1_v,
                                      b2_g, b2_b, b2_m, b2_v, pq, pk);
  simaff_kernel<<<1024, 256, 0, stream>>>(pq, pk, aff);
  pv_kernel<<<4096, 256, 0, stream>>>(y, Wvc, b3_g, b3_b, b3_m, b3_v, pv);
  out_kernel<<<4096, 256, 0, stream>>>(aff, pv, y, alpha, out);
}
